// MultiHeadAttention_73650099192109
// MI455X (gfx1250) — compile-verified, matrix operand order checked by reading, not run on hardware
//
#include <hip/hip_runtime.h>

// Problem constants (match reference)
#define BB 4
#define TT 2048
#define EE 1024
#define HH 16
#define HD 64

typedef __attribute__((ext_vector_type(16))) _Float16 v16h;
typedef __attribute__((ext_vector_type(8)))  _Float16 v8h;
typedef __attribute__((ext_vector_type(8)))  float    v8f;

__device__ __forceinline__ v8f wmma32(v16h a, v16h b, v8f c) {
  // D = A(16x32 f16) * B(32x16 f16) + C(16x16 f32)
  return __builtin_amdgcn_wmma_f32_16x16x32_f16(
      /*neg_a=*/false, a, /*neg_b=*/false, b,
      /*c_mod=*/(short)0, c, /*reuse_a=*/false, /*reuse_b=*/false);
}

// ---- fragment loader (CDNA5 wave32 WMMA layouts, ISA 7.12.2) ----
// Works for A (row = lane&15, K pattern) and for n-major B (col = lane&15,
// contiguous k). Two aligned 16-byte loads per lane.
__device__ __forceinline__ v16h load_frag16(const _Float16* p) {
  const v8h lo = *(const v8h*)(p);
  const v8h hi = *(const v8h*)(p + 16);
  return __builtin_shufflevector(lo, hi, 0, 1, 2, 3, 4, 5, 6, 7, 8, 9, 10, 11,
                                 12, 13, 14, 15);
}
__device__ __forceinline__ v16h load_mat(const _Float16* base, int ld,
                                         int lane) {
  return load_frag16(base + (size_t)(lane & 15) * ld + ((lane >> 4) << 3));
}

// ================= prep kernels (one-shot f32->f16 + layout) =================
__global__ __launch_bounds__(256) void cvt_f32_f16_kernel(
    const float* __restrict__ in, _Float16* __restrict__ out, int n) {
  int i = (int)(blockIdx.x * blockDim.x + threadIdx.x);
  if (i < n) out[i] = (_Float16)in[i];
}

// W[mat][h][e][d] (f32) -> Wt[((mat*HH+h)*HD + d)*EE + e] (f16, n-major)
__global__ __launch_bounds__(256) void wqkv_prep_kernel(
    const float* __restrict__ Wq, const float* __restrict__ Wk,
    const float* __restrict__ Wv, _Float16* __restrict__ wt) {
  int i = (int)(blockIdx.x * blockDim.x + threadIdx.x);
  int e = i % EE;
  int r = i / EE;  // r = (mat*HH + h)*HD + d
  if (r >= 3 * HH * HD) return;
  int d = r % HD;
  int mh = r / HD;
  int h = mh % HH;
  int mat = mh / HH;
  const float* W = (mat == 0 ? Wq : (mat == 1 ? Wk : Wv));
  wt[(size_t)r * EE + e] = (_Float16)W[((size_t)h * EE + e) * HD + d];
}

// ---------------- Kernel 1: per-head Q/K/V projections ----------------
// one wave -> 16(t) x 64(d) tile of one of q/k/v for one (b, h)
// q,k stored [b][h][t][d]; v stored transposed [b][h][d][t]
__global__ __launch_bounds__(256) void qkv_proj_kernel(
    const _Float16* __restrict__ xh, const _Float16* __restrict__ wt,
    _Float16* __restrict__ qb, _Float16* __restrict__ kb,
    _Float16* __restrict__ vtb) {
  const int lane = threadIdx.x & 31;
  const int gw = (int)((blockIdx.x * blockDim.x + threadIdx.x) >> 5);
  const int TT16 = TT / 16;
  const int total = 3 * HH * BB * TT16;
  if (gw >= total) return;
  int ttile = gw % TT16;
  int rest = gw / TT16;
  int b = rest % BB;
  rest /= BB;
  int h = rest % HH;
  int mat = rest / HH;
  const int t0 = ttile * 16;

  const _Float16* W = wt + (size_t)(mat * HH + h) * HD * EE;  // [d][e] n-major
  const _Float16* xp = xh + ((size_t)b * TT + t0) * EE;

  v8f C0 = {}, C1 = {}, C2 = {}, C3 = {};
  for (int kk = 0; kk < EE; kk += 32) {
    v16h A = load_mat(xp + kk, EE, lane);
    v16h B0 = load_mat(W + (size_t)0 * 16 * EE + kk, EE, lane);
    v16h B1 = load_mat(W + (size_t)1 * 16 * EE + kk, EE, lane);
    v16h B2 = load_mat(W + (size_t)2 * 16 * EE + kk, EE, lane);
    v16h B3 = load_mat(W + (size_t)3 * 16 * EE + kk, EE, lane);
    C0 = wmma32(A, B0, C0);
    C1 = wmma32(A, B1, C1);
    C2 = wmma32(A, B2, C2);
    C3 = wmma32(A, B3, C3);
  }
  const int hi = lane >> 4, c = lane & 15;
  if (mat < 2) {
    _Float16* outp =
        (mat == 0 ? qb : kb) + ((size_t)(b * HH + h) * TT + t0) * HD;
#pragma unroll
    for (int i = 0; i < 8; ++i) {
      int r = i + 8 * hi;
      outp[r * HD + 0 + c] = (_Float16)C0[i];
      outp[r * HD + 16 + c] = (_Float16)C1[i];
      outp[r * HD + 32 + c] = (_Float16)C2[i];
      outp[r * HD + 48 + c] = (_Float16)C3[i];
    }
  } else {
    // transposed store: vT[b][h][d][t]; per lane 8 consecutive t -> one b128
    _Float16* vtp = vtb + (size_t)(b * HH + h) * HD * TT;
    v8h s0v, s1v, s2v, s3v;
#pragma unroll
    for (int i = 0; i < 8; ++i) {
      s0v[i] = (_Float16)C0[i];
      s1v[i] = (_Float16)C1[i];
      s2v[i] = (_Float16)C2[i];
      s3v[i] = (_Float16)C3[i];
    }
    size_t tofs = (size_t)t0 + 8 * hi;
    *(v8h*)(vtp + (size_t)(0 + c) * TT + tofs) = s0v;
    *(v8h*)(vtp + (size_t)(16 + c) * TT + tofs) = s1v;
    *(v8h*)(vtp + (size_t)(32 + c) * TT + tofs) = s2v;
    *(v8h*)(vtp + (size_t)(48 + c) * TT + tofs) = s3v;
  }
}

// ---------------- Kernel 2: causal flash attention ----------------
// one wave -> 16 query rows for one (b, h); online softmax over 32-key blocks
__global__ __launch_bounds__(256) void flash_attn_kernel(
    const _Float16* __restrict__ qb, const _Float16* __restrict__ kb,
    const _Float16* __restrict__ vtb, _Float16* __restrict__ attnb) {
  __shared__ alignas(16) _Float16 plds[8][16 * 32];  // per-wave P staging
  const int lane = threadIdx.x & 31;
  const int widx = threadIdx.x >> 5;
  const int gw = (int)((blockIdx.x * blockDim.x + threadIdx.x) >> 5);
  const int TT16 = TT / 16;
  const int total = BB * HH * TT16;
  if (gw >= total) return;
  int ttile = gw % TT16;
  int bh = gw / TT16;
  int h = bh % HH;
  int b = bh / HH;
  const int t0 = ttile * 16;

  const _Float16* qp = qb + (size_t)(b * HH + h) * TT * HD;
  const _Float16* kp = kb + (size_t)(b * HH + h) * TT * HD;
  const _Float16* vtp = vtb + (size_t)(b * HH + h) * HD * TT;  // [d][t]

  // Q tile as two A fragments (d = 0..31, 32..63)
  v16h QA0 = load_mat(qp + (size_t)t0 * HD + 0, HD, lane);
  v16h QA1 = load_mat(qp + (size_t)t0 * HD + 32, HD, lane);

  v8f O0 = {}, O1 = {}, O2 = {}, O3 = {};
  float m[8], l[8];
#pragma unroll
  for (int i = 0; i < 8; ++i) {
    m[i] = -__builtin_inff();
    l[i] = 0.0f;
  }

  const int hi = lane >> 4, c = lane & 15;
  const float scale = 0.125f;  // 1/sqrt(HD)

  for (int s0 = 0; s0 < t0 + 16; s0 += 32) {
    // ---- S = (Q K^T) for keys [s0, s0+32) : 4 WMMAs ----
    // B fragment is n-major over keys: K[t][d] rows are contiguous in d.
    v8f S0 = {}, S1 = {};
    {
      const _Float16* k0 = kp + (size_t)s0 * HD;
      const _Float16* k1 = kp + (size_t)(s0 + 16) * HD;
      v16h KB00 = load_mat(k0 + 0, HD, lane);
      v16h KB01 = load_mat(k0 + 32, HD, lane);
      v16h KB10 = load_mat(k1 + 0, HD, lane);
      v16h KB11 = load_mat(k1 + 32, HD, lane);
      S0 = wmma32(QA0, KB00, S0);
      S0 = wmma32(QA1, KB01, S0);
      S1 = wmma32(QA0, KB10, S1);
      S1 = wmma32(QA1, KB11, S1);
    }
    // prefetch next key block (gfx1250 global_prefetch_b8)
    __builtin_prefetch(kp + (size_t)(s0 + 32) * HD, 0, 1);
    __builtin_prefetch(vtp + (size_t)s0 + 32, 0, 1);

    // ---- causal mask + block row-max (rows live across 16-lane halves) ----
    const int col0 = s0 + c, col1 = col0 + 16;
    float mb[8];
#pragma unroll
    for (int i = 0; i < 8; ++i) {
      int r = t0 + i + 8 * hi;  // global query row
      float a0 = (col0 <= r) ? S0[i] * scale : -__builtin_inff();
      float a1 = (col1 <= r) ? S1[i] * scale : -__builtin_inff();
      S0[i] = a0;
      S1[i] = a1;
      mb[i] = fmaxf(a0, a1);
    }
#pragma unroll
    for (int off = 1; off < 16; off <<= 1) {
#pragma unroll
      for (int i = 0; i < 8; ++i)
        mb[i] = fmaxf(mb[i], __shfl_xor(mb[i], off, 32));
    }

    // ---- online softmax update ----
    float rs[8];
#pragma unroll
    for (int i = 0; i < 8; ++i) {
      float mn = fmaxf(m[i], mb[i]);
      float al = __expf(m[i] - mn);
      m[i] = mn;
      float p0 = __expf(S0[i] - mn);
      float p1 = __expf(S1[i] - mn);
      S0[i] = p0;
      S1[i] = p1;
      rs[i] = p0 + p1;
      O0[i] *= al;
      O1[i] *= al;
      O2[i] *= al;
      O3[i] *= al;
      l[i] *= al;
    }
#pragma unroll
    for (int off = 1; off < 16; off <<= 1) {
#pragma unroll
      for (int i = 0; i < 8; ++i) rs[i] += __shfl_xor(rs[i], off, 32);
    }
#pragma unroll
    for (int i = 0; i < 8; ++i) l[i] += rs[i];

    // ---- P: C-layout -> A-layout via per-wave LDS tile (DS in-order/wave) ----
    _Float16* pl = plds[widx];
#pragma unroll
    for (int i = 0; i < 8; ++i) {
      int r = i + 8 * hi;
      pl[r * 32 + c] = (_Float16)S0[i];
      pl[r * 32 + 16 + c] = (_Float16)S1[i];
    }
    v16h P = load_mat(pl, 32, lane);

    // ---- O += P * V : 4 WMMAs; V^T is n-major over d, contiguous in t ----
    const _Float16* vbase = vtp + (size_t)s0;
    v16h V0 = load_mat(vbase + (size_t)0 * 16 * TT, TT, lane);
    v16h V1 = load_mat(vbase + (size_t)1 * 16 * TT, TT, lane);
    v16h V2 = load_mat(vbase + (size_t)2 * 16 * TT, TT, lane);
    v16h V3 = load_mat(vbase + (size_t)3 * 16 * TT, TT, lane);
    O0 = wmma32(P, V0, O0);
    O1 = wmma32(P, V1, O1);
    O2 = wmma32(P, V2, O2);
    O3 = wmma32(P, V3, O3);
  }

  // ---- normalize and store heads concatenated: attn[b][t][h*64 + d] ----
  _Float16* ap = attnb + ((size_t)b * TT + t0) * EE + h * HD;
#pragma unroll
  for (int i = 0; i < 8; ++i) {
    float inv = 1.0f / l[i];
    int r = i + 8 * hi;
    ap[(size_t)r * EE + 0 + c] = (_Float16)(O0[i] * inv);
    ap[(size_t)r * EE + 16 + c] = (_Float16)(O1[i] * inv);
    ap[(size_t)r * EE + 32 + c] = (_Float16)(O2[i] * inv);
    ap[(size_t)r * EE + 48 + c] = (_Float16)(O3[i] * inv);
  }
}

// ------------- Kernel 3: output projection  out = attn @ Wo^T + bo -------------
__global__ __launch_bounds__(256) void out_proj_kernel(
    const _Float16* __restrict__ attnb, const _Float16* __restrict__ woh,
    const float* __restrict__ bo, float* __restrict__ out) {
  const int lane = threadIdx.x & 31;
  const int gw = (int)((blockIdx.x * blockDim.x + threadIdx.x) >> 5);
  const int RT = (BB * TT) / 16;  // 512 row tiles
  const int CT = EE / 64;         // 16 col groups
  if (gw >= RT * CT) return;
  int ct = gw % CT;
  int rt = gw / CT;
  const int row0 = rt * 16, n0 = ct * 64;

  v8f C0 = {}, C1 = {}, C2 = {}, C3 = {};
  for (int kk = 0; kk < EE; kk += 32) {
    v16h A = load_mat(attnb + (size_t)row0 * EE + kk, EE, lane);
    // out[m][n] = sum_k attn[m][k] * Wo[n][k]  -> rows of Wo, contiguous in k
    v16h B0 = load_mat(woh + (size_t)(n0 + 0) * EE + kk, EE, lane);
    v16h B1 = load_mat(woh + (size_t)(n0 + 16) * EE + kk, EE, lane);
    v16h B2 = load_mat(woh + (size_t)(n0 + 32) * EE + kk, EE, lane);
    v16h B3 = load_mat(woh + (size_t)(n0 + 48) * EE + kk, EE, lane);
    C0 = wmma32(A, B0, C0);
    C1 = wmma32(A, B1, C1);
    C2 = wmma32(A, B2, C2);
    C3 = wmma32(A, B3, C3);
  }
  const int hi = lane >> 4, c = lane & 15;
#pragma unroll
  for (int i = 0; i < 8; ++i) {
    int r = row0 + i + 8 * hi;
    out[(size_t)r * EE + n0 + 0 + c] = C0[i] + bo[n0 + 0 + c];
    out[(size_t)r * EE + n0 + 16 + c] = C1[i] + bo[n0 + 16 + c];
    out[(size_t)r * EE + n0 + 32 + c] = C2[i] + bo[n0 + 32 + c];
    out[(size_t)r * EE + n0 + 48 + c] = C3[i] + bo[n0 + 48 + c];
  }
}

extern "C" void kernel_launch(void* const* d_in, const int* in_sizes, int n_in,
                              void* d_out, int out_size, void* d_ws,
                              size_t ws_size, hipStream_t stream) {
  (void)in_sizes; (void)n_in; (void)out_size; (void)ws_size;
  const float* x = (const float*)d_in[0];
  const float* Wq = (const float*)d_in[1];
  const float* Wk = (const float*)d_in[2];
  const float* Wv = (const float*)d_in[3];
  const float* Wo = (const float*)d_in[4];
  const float* bo = (const float*)d_in[5];
  float* out = (float*)d_out;

  // workspace layout (f16 elements), ~92 MB total
  const size_t NX = (size_t)BB * TT * EE;       // x / attn
  const size_t NWT = (size_t)3 * HH * HD * EE;  // transposed qkv weights
  const size_t NWO = (size_t)EE * EE;           // Wo f16
  const size_t NQ = (size_t)BB * HH * TT * HD;  // q / k / vT
  _Float16* xh = (_Float16*)d_ws;
  _Float16* wt = xh + NX;
  _Float16* woh = wt + NWT;
  _Float16* qb = woh + NWO;
  _Float16* kb = qb + NQ;
  _Float16* vtb = kb + NQ;
  _Float16* ab = vtb + NQ;

  const int block = 256;
  // ---- prep: conversions / transposes (tiny vs total traffic) ----
  cvt_f32_f16_kernel<<<(int)((NX + block - 1) / block), block, 0, stream>>>(
      x, xh, (int)NX);
  cvt_f32_f16_kernel<<<(int)((NWO + block - 1) / block), block, 0, stream>>>(
      Wo, woh, (int)NWO);
  wqkv_prep_kernel<<<(int)((NWT + block - 1) / block), block, 0, stream>>>(
      Wq, Wk, Wv, wt);

  {
    const int waves = 3 * HH * BB * (TT / 16);
    const long long threads = (long long)waves * 32;
    qkv_proj_kernel<<<(int)((threads + block - 1) / block), block, 0, stream>>>(
        xh, wt, qb, kb, vtb);
  }
  {
    const int waves = BB * HH * (TT / 16);
    const long long threads = (long long)waves * 32;
    flash_attn_kernel<<<(int)((threads + block - 1) / block), block, 0,
                        stream>>>(qb, kb, vtb, ab);
  }
  {
    const int waves = (BB * TT / 16) * (EE / 64);
    const long long threads = (long long)waves * 32;
    out_proj_kernel<<<(int)((threads + block - 1) / block), block, 0, stream>>>(
        ab, Wo ? woh : woh, bo, out);
  }
}
